// MultiHeadAttention_47175920779443
// MI455X (gfx1250) — matrix operand order repaired and checked by reading, not compiled, not run on hardware
//
#include <hip/hip_runtime.h>

// Problem constants (from reference): B=2, T=2048, HID=1024, HEADS=16
#define BSZ   2
#define TSEQ  2048
#define HIDD  1024
#define HEADS 16
#define HD    64
#define N3    (3 * HIDD)
#define BT    (BSZ * TSEQ)

typedef __bf16 bf16_t;
typedef __attribute__((ext_vector_type(8)))  __bf16 v8bf;
typedef __attribute__((ext_vector_type(16))) __bf16 v16bf;
typedef __attribute__((ext_vector_type(8)))  float  v8f;
typedef __attribute__((ext_vector_type(4)))  unsigned int v4u;
typedef __attribute__((ext_vector_type(8)))  int v8i;
typedef __attribute__((ext_vector_type(4)))  int v4i;

__device__ __forceinline__ v16bf cat8(v8bf lo, v8bf hi) {
  return __builtin_shufflevector(lo, hi, 0,1,2,3,4,5,6,7,8,9,10,11,12,13,14,15);
}

__device__ __forceinline__ v8f wmma_bf16(v16bf a, v16bf b, v8f c) {
  // 8-arg form: (neg_a, A, neg_b, B, c_mod, C, reuse_a, reuse_b)
  return __builtin_amdgcn_wmma_f32_16x16x32_bf16(false, a, false, b, (short)0, c,
                                                 false, false);
}

// ---------------------------------------------------------------------------
// TDM: issue a 2D bf16 tile load (tile_d0 x tile_d1, row-major) into LDS.
// D# packing per CDNA5 ISA ch.8 (groups 0/1; groups 2/3 zero => 2D tensor).
// Tracked by TENSORcnt; completion via s_wait_tensorcnt.
// This toolchain exposes the 6-arg builtin:
//   (uint32x4 g0, int32x8 g1, int32x4, int32x4, int32x8, i32 cpol)
// ---------------------------------------------------------------------------
__device__ __forceinline__ void tdm_load_2d(unsigned lds_off, const bf16_t* gptr,
                                            unsigned tensor_d0, unsigned tensor_d1,
                                            unsigned tile_d0, unsigned tile_d1,
                                            unsigned stride0) {
  const unsigned long long ga = (unsigned long long)(uintptr_t)gptr;
  v4u g0;
  g0[0] = 1u;                                            // count=1 (user D#)
  g0[1] = lds_off;                                       // lds_addr [63:32]
  g0[2] = (unsigned)(ga & 0xffffffffu);                  // global_addr [95:64]
  g0[3] = (unsigned)((ga >> 32) & 0x01ffffffu)           // global_addr [120:96]
        | (2u << 30);                                    // type=2 ("image")
  v8i g1;
  g1[0] = (int)(1u << 16);                               // data_size=1 (2B bf16)
  g1[1] = (int)((tensor_d0 & 0xffffu) << 16);            // tensor_dim0[15:0]
  g1[2] = (int)(((tensor_d0 >> 16) & 0xffffu)
        | ((tensor_d1 & 0xffffu) << 16));                // dim0[31:16] | dim1[15:0]
  g1[3] = (int)(((tensor_d1 >> 16) & 0xffffu)
        | (tile_d0 << 16));                              // dim1[31:16] | tile_dim0
  g1[4] = (int)tile_d1;                                  // tile_dim1 | tile_dim2=0
  g1[5] = (int)stride0;                                  // tensor_dim0_stride[31:0]
  g1[6] = 0;                                             // stride0[47:32] | stride1 lo
  g1[7] = 0;                                             // stride1 hi
  const v4i z4 = {0, 0, 0, 0};
  const v8i z8 = {0, 0, 0, 0, 0, 0, 0, 0};
  __builtin_amdgcn_tensor_load_to_lds(g0, g1, z4, z4, z8, 0);
}

// ---------------------------------------------------------------------------
// Kernel 1: fp32 -> bf16 conversion; W transposed once:  wt[n][k] == W^T
// ---------------------------------------------------------------------------
__global__ void cvt_kernel(const float* __restrict__ x, const float* __restrict__ W,
                           bf16_t* __restrict__ xb, bf16_t* __restrict__ wt) {
  const int n1 = BT * HIDD;
  const int n2 = HIDD * N3;
  for (int i = blockIdx.x * blockDim.x + threadIdx.x; i < n1 + n2;
       i += gridDim.x * blockDim.x) {
    if (i < n1) {
      xb[i] = (bf16_t)x[i];
    } else {
      const int j = i - n1;
      const int k = j / N3;
      const int n = j - k * N3;
      wt[(size_t)n * HIDD + k] = (bf16_t)W[j];     // transpose
    }
  }
}

// ---------------------------------------------------------------------------
// Kernel 2: qkv = (x @ W) * bias   (bias is MULTIPLICATIVE per reference)
// Tile: 128(M) x 128(N), K-step 64, 8 waves, wave tile 32(M) x 64(N).
// Tile staging by the Tensor Data Mover, double-buffered in LDS:
//   wave0 issues DMA for tile i+1, s_wait_tensorcnt(2) => tile i resident.
// Epilogue scatters into attention layouts:
//   q,k : [b,h,t,d]     vT : [b,h,d,t]
// Column o decomposes as o = d*48 + kq*16 + h  (reshape (d,3,h)).
// ---------------------------------------------------------------------------
__global__ __launch_bounds__(256)
void qkv_kernel(const bf16_t* __restrict__ xb, const bf16_t* __restrict__ wt,
                const float* __restrict__ bias,
                bf16_t* __restrict__ qb, bf16_t* __restrict__ kbuf,
                bf16_t* __restrict__ vt) {
  __shared__ alignas(32) bf16_t sA[2][128 * 64];   // [m][k], double buffered
  __shared__ alignas(32) bf16_t sW[2][128 * 64];   // [n][k], double buffered

  const int tid  = threadIdx.x;
  const int lane = tid & 31;
  const int l15  = lane & 15;
  const int half = lane >> 4;
  const int w    = tid >> 5;
  const int wm   = w >> 1;              // 0..3 : 32-row M strip
  const int wn   = w & 1;               // 0..1 : 64-col N strip
  const int blockN = blockIdx.x * 128;
  const int blockM = blockIdx.y * 128;

  const bf16_t* gA = xb + (size_t)blockM * HIDD;   // A tile origin (k advances)
  const bf16_t* gB = wt + (size_t)blockN * HIDD;   // B tile origin (k advances)

  const v8f vzero = {0.f,0.f,0.f,0.f,0.f,0.f,0.f,0.f};
  v8f acc[2][4];
  #pragma unroll
  for (int mt = 0; mt < 2; ++mt)
    #pragma unroll
    for (int nt = 0; nt < 4; ++nt) acc[mt][nt] = vzero;

  // prologue: DMA tile 0 into buffer 0 (wave 0 only; TDM is per-wave)
  if (w == 0) {
    tdm_load_2d((unsigned)(uintptr_t)&sA[0][0], gA, HIDD, BT, 64, 128, HIDD);
    tdm_load_2d((unsigned)(uintptr_t)&sW[0][0], gB, HIDD, N3, 64, 128, HIDD);
  }

  for (int k0 = 0; k0 < HIDD; k0 += 64) {
    const int cur = (k0 >> 6) & 1;
    if (w == 0) {
      if (k0 + 64 < HIDD) {
        // DMA next tile into the other buffer, then wait for current tile:
        // TENSORcnt is in-order, so cnt<=2 => the two older ops completed.
        tdm_load_2d((unsigned)(uintptr_t)&sA[cur ^ 1][0], gA + (k0 + 64),
                    HIDD, BT, 64, 128, HIDD);
        tdm_load_2d((unsigned)(uintptr_t)&sW[cur ^ 1][0], gB + (k0 + 64),
                    HIDD, N3, 64, 128, HIDD);
        __builtin_amdgcn_s_wait_tensorcnt(2);
      } else {
        __builtin_amdgcn_s_wait_tensorcnt(0);
      }
    }
    __syncthreads();   // current buffer resident for all 8 waves

    // A fragments: lane<16 -> rows M, K in {kb..kb+7} U {kb+16..kb+23}
    v16bf afr[2][2];
    #pragma unroll
    for (int mt = 0; mt < 2; ++mt) {
      const int row = wm * 32 + mt * 16 + l15;
      #pragma unroll
      for (int kk = 0; kk < 2; ++kk) {
        const int kb2 = kk * 32 + half * 8;
        v8bf lo = *(const v8bf*)&sA[cur][row * 64 + kb2];
        v8bf hi = *(const v8bf*)&sA[cur][row * 64 + kb2 + 16];
        afr[mt][kk] = cat8(lo, hi);
      }
    }
    // B fragments: lane -> column N, K in {kb..kb+7} U {kb+16..kb+23} (kb = half*8)
    #pragma unroll
    for (int nt = 0; nt < 4; ++nt) {
      const int n = wn * 64 + nt * 16 + l15;
      #pragma unroll
      for (int kk = 0; kk < 2; ++kk) {
        v16bf bfr = cat8(*(const v8bf*)&sW[cur][n * 64 + kk * 32 + half * 8], *(const v8bf*)&sW[cur][n * 64 + kk * 32 + half * 8 + 16]);
        #pragma unroll
        for (int mt = 0; mt < 2; ++mt)
          acc[mt][nt] = wmma_bf16(afr[mt][kk], bfr, acc[mt][nt]);
      }
    }
    __syncthreads();   // all reads of buffer `cur` done before its next DMA
  }

  // Epilogue: multiplicative bias + scatter to q / k / vT (bf16)
  #pragma unroll
  for (int mt = 0; mt < 2; ++mt) {
    #pragma unroll
    for (int nt = 0; nt < 4; ++nt) {
      #pragma unroll
      for (int r = 0; r < 8; ++r) {
        const int m = blockM + wm * 32 + mt * 16 + r + half * 8;
        const int n = blockN + wn * 64 + nt * 16 + l15;
        const float v = acc[mt][nt][r] * bias[n];
        const int h  = n & 15;
        const int kq = (n >> 4) % 3;
        const int d  = n / 48;
        const int bi = m >> 11;       // m / 2048
        const int t  = m & 2047;
        const int bh = bi * HEADS + h;
        const bf16_t bv = (bf16_t)v;
        if (kq == 0)      qb  [((size_t)bh * TSEQ + t) * HD + d] = bv;
        else if (kq == 1) kbuf[((size_t)bh * TSEQ + t) * HD + d] = bv;
        else              vt  [((size_t)bh * HD + d) * TSEQ + t] = bv;
      }
    }
  }
}

// ---------------------------------------------------------------------------
// Kernel 3: flash attention per (b,h). Wave owns 16 q rows.
// S = (Q K^T) / 32, online softmax, O += P V. Output fp32 [b,t,h*64+d].
// ---------------------------------------------------------------------------
__global__ __launch_bounds__(256)
void attn_kernel(const bf16_t* __restrict__ qb, const bf16_t* __restrict__ kbuf,
                 const bf16_t* __restrict__ vt, float* __restrict__ out) {
  __shared__ alignas(32) bf16_t sP[8 * 16 * 64];   // per-wave P staging

  const int tid  = threadIdx.x;
  const int lane = tid & 31;
  const int l15  = lane & 15;
  const int half = lane >> 4;
  const int w    = tid >> 5;
  const int bh   = blockIdx.y;
  const int bi   = bh >> 4;
  const int h    = bh & 15;
  const int t0   = blockIdx.x * 128 + w * 16;      // this wave's q rows

  const bf16_t* qbase = qb   + (size_t)bh * TSEQ * HD;
  const bf16_t* kbase = kbuf + (size_t)bh * TSEQ * HD;
  const bf16_t* vbase = vt   + (size_t)bh * HD * TSEQ;
  bf16_t* pw = &sP[w * 16 * 64];

  // Q A-fragments for d-chunks {0..31},{32..63}
  v16bf aq[2];
  {
    const bf16_t* qp = qbase + (size_t)(t0 + l15) * HD;
    #pragma unroll
    for (int kk = 0; kk < 2; ++kk) {
      const int kb2 = kk * 32 + half * 8;
      v8bf lo = *(const v8bf*)(qp + kb2);
      v8bf hi = *(const v8bf*)(qp + kb2 + 16);
      aq[kk] = cat8(lo, hi);
    }
  }

  const v8f vzero = {0.f,0.f,0.f,0.f,0.f,0.f,0.f,0.f};
  v8f oacc[4];
  #pragma unroll
  for (int dt = 0; dt < 4; ++dt) oacc[dt] = vzero;
  float mstate[8], lstate[8];
  #pragma unroll
  for (int r = 0; r < 8; ++r) { mstate[r] = -3.0e38f; lstate[r] = 0.f; }

  for (int kb0 = 0; kb0 < TSEQ; kb0 += 64) {
    // ---- S = Q K^T for a 16x64 strip ----
    v8f s[4];
    #pragma unroll
    for (int nt = 0; nt < 4; ++nt) s[nt] = vzero;
    #pragma unroll
    for (int kk = 0; kk < 2; ++kk) {
      const int dd = kk * 32 + half * 8;
      #pragma unroll
      for (int nt = 0; nt < 4; ++nt) {
        const int key = kb0 + nt * 16 + l15;
        v16bf bk = cat8(*(const v8bf*)(kbase + (size_t)key * HD + dd), *(const v8bf*)(kbase + (size_t)key * HD + dd + 16));
        s[nt] = wmma_bf16(aq[kk], bk, s[nt]);
      }
    }
    // prefetch next key block (K and V)
    if (kb0 + 64 < TSEQ) {
      __builtin_prefetch(kbase + (size_t)(kb0 + 64 + l15) * HD, 0, 1);
      __builtin_prefetch(vbase + (size_t)l15 * TSEQ + kb0 + 64, 0, 1);
    }
    // ---- online softmax (scale 1/sqrt(1024) = 1/32) ----
    #pragma unroll
    for (int r = 0; r < 8; ++r) {
      float v0 = s[0][r] * 0.03125f, v1 = s[1][r] * 0.03125f;
      float v2 = s[2][r] * 0.03125f, v3 = s[3][r] * 0.03125f;
      float rm = fmaxf(fmaxf(v0, v1), fmaxf(v2, v3));
      rm = fmaxf(rm, __shfl_xor(rm, 1, 32));
      rm = fmaxf(rm, __shfl_xor(rm, 2, 32));
      rm = fmaxf(rm, __shfl_xor(rm, 4, 32));
      rm = fmaxf(rm, __shfl_xor(rm, 8, 32));
      const float newm = fmaxf(mstate[r], rm);
      const float corr = __expf(mstate[r] - newm);
      mstate[r] = newm;
      const float p0 = __expf(v0 - newm), p1 = __expf(v1 - newm);
      const float p2 = __expf(v2 - newm), p3 = __expf(v3 - newm);
      float rs = p0 + p1 + p2 + p3;
      rs += __shfl_xor(rs, 1, 32);
      rs += __shfl_xor(rs, 2, 32);
      rs += __shfl_xor(rs, 4, 32);
      rs += __shfl_xor(rs, 8, 32);
      lstate[r] = lstate[r] * corr + rs;
      #pragma unroll
      for (int dt = 0; dt < 4; ++dt) oacc[dt][r] *= corr;
      // stage P (bf16) to wave-private LDS for C-layout -> A-layout
      bf16_t* prow = pw + (r + half * 8) * 64;
      prow[l15]      = (bf16_t)p0;
      prow[16 + l15] = (bf16_t)p1;
      prow[32 + l15] = (bf16_t)p2;
      prow[48 + l15] = (bf16_t)p3;
    }
    // ---- P A-fragments (DS ops are in-order within a wave) ----
    v16bf ap[2];
    #pragma unroll
    for (int kk = 0; kk < 2; ++kk) {
      const bf16_t* pr = pw + l15 * 64 + kk * 32 + half * 8;
      v8bf lo = *(const v8bf*)pr;
      v8bf hi = *(const v8bf*)(pr + 16);
      ap[kk] = cat8(lo, hi);
    }
    // ---- O += P V (v stored [b,h,d,t] so B-frags are contiguous over t) ----
    #pragma unroll
    for (int kk = 0; kk < 2; ++kk) {
      const int tb = kb0 + kk * 32 + half * 8;
      #pragma unroll
      for (int dt = 0; dt < 4; ++dt) {
        const int d = dt * 16 + l15;
        v16bf bv = cat8(*(const v8bf*)(vbase + (size_t)d * TSEQ + tb), *(const v8bf*)(vbase + (size_t)d * TSEQ + tb + 16));
        oacc[dt] = wmma_bf16(ap[kk], bv, oacc[dt]);
      }
    }
  }

  // ---- finalize: divide by l, write fp32 out[b, t, h*64 + d] ----
  #pragma unroll
  for (int r = 0; r < 8; ++r) {
    const int tg = t0 + r + half * 8;
    const float invl = 1.0f / lstate[r];
    float* op = out + ((size_t)(bi * TSEQ + tg)) * HIDD + h * HD;
    #pragma unroll
    for (int dt = 0; dt < 4; ++dt)
      op[dt * 16 + l15] = oacc[dt][r] * invl;
  }
}

// ---------------------------------------------------------------------------
extern "C" void kernel_launch(void* const* d_in, const int* in_sizes, int n_in,
                              void* d_out, int out_size, void* d_ws, size_t ws_size,
                              hipStream_t stream) {
  (void)in_sizes; (void)n_in; (void)out_size; (void)ws_size;
  const float* x    = (const float*)d_in[0];
  const float* W    = (const float*)d_in[1];
  const float* bias = (const float*)d_in[2];
  // d_in[3] = num_heads (compile-time constant HEADS)

  char* ws = (char*)d_ws;
  bf16_t* xb   = (bf16_t*)(ws + (size_t)0);                  //  8 MiB
  bf16_t* wt   = (bf16_t*)(ws + (size_t)8  * 1024 * 1024);   //  6 MiB (W^T)
  bf16_t* qb   = (bf16_t*)(ws + (size_t)14 * 1024 * 1024);   //  8 MiB
  bf16_t* kbuf = (bf16_t*)(ws + (size_t)22 * 1024 * 1024);   //  8 MiB
  bf16_t* vt   = (bf16_t*)(ws + (size_t)30 * 1024 * 1024);   //  8 MiB

  cvt_kernel<<<2048, 256, 0, stream>>>(x, W, xb, wt);

  dim3 g2(N3 / 128, BT / 128);      // (24, 32)
  qkv_kernel<<<g2, 256, 0, stream>>>(xb, wt, bias, qb, kbuf, vt);

  dim3 g3(TSEQ / 128, BSZ * HEADS); // (16, 32)
  attn_kernel<<<g3, 256, 0, stream>>>(qb, kbuf, vt, (float*)d_out);
}
